// PAM_26912265077350
// MI455X (gfx1250) — compile-verified, matrix operand order checked by reading, not run on hardware
//
#include <hip/hip_runtime.h>

// PAM attention, MI455X / gfx1250, wave32, WMMA f16->f32.
// B=8, C=128, H=W=64, N=4096.
#define BATCH 8
#define CCH   128
#define NPOS  4096   // H*W

typedef __attribute__((ext_vector_type(16))) _Float16 v16h;
typedef __attribute__((ext_vector_type(8)))  float    v8f;

// 16x32 f16 fragment = 16 halfs/lane = 2 x 16-byte chunks (K 0..7+g and 16..23+g)
union Frag16 { v16h h; uint4 q[2]; };

// ---------------------------------------------------------------------------
// K1: channel softmax. xs (f32) -> d_out part 1; V[c][n] and Vt[n][c] as f16.
// One thread per (b, n); x reads coalesced across n for each c; x is 16 MB ->
// fully L2 resident, so the 3 passes over c cost nothing at HBM.
// ---------------------------------------------------------------------------
__global__ void pam_channel_softmax(const float* __restrict__ x,
                                    float* __restrict__ xs_out,
                                    _Float16* __restrict__ V,
                                    _Float16* __restrict__ Vt) {
    const int n = blockIdx.x * blockDim.x + threadIdx.x;   // 0..N-1
    const int b = blockIdx.y;
    const size_t base  = (size_t)b * CCH * NPOS + n;       // stride NPOS over c
    float m = -3.4e38f;
    for (int c = 0; c < CCH; ++c) m = fmaxf(m, x[base + (size_t)c * NPOS]);
    float s = 0.f;
    for (int c = 0; c < CCH; ++c) s += __expf(x[base + (size_t)c * NPOS] - m);
    const float inv = 1.f / s;
    const size_t tbase = ((size_t)b * NPOS + n) * CCH;
    for (int c = 0; c < CCH; ++c) {
        const float e = __expf(x[base + (size_t)c * NPOS] - m) * inv;
        xs_out[base + (size_t)c * NPOS] = e;
        const _Float16 eh = (_Float16)e;
        V [base + (size_t)c * NPOS] = eh;   // V[b][c][n]  (contig in n=m, for GEMM-2 A)
        Vt[tbase + c]               = eh;   // Vt[b][n][c] (contig in c=K, for GEMM-1 A/B)
    }
}

// ---------------------------------------------------------------------------
// K2: energy = V^T V per batch, streamed raw into the attention slab.
// Workgroup = (b, 16-row block). 8 waves x 32 tiles of 16x16, K=C=128 as
// 4 x v_wmma_f32_16x16x32_f16. A fragments (shared by all tiles) held in
// registers; all operand loads are 2 contiguous 16B loads/lane from Vt
// (L2-resident, 1 MB/batch).
// ---------------------------------------------------------------------------
__global__ void pam_energy(const _Float16* __restrict__ Vt,
                           float* __restrict__ energy) {
    const int b    = blockIdx.y;
    const int n0   = blockIdx.x * 16;
    const int lane = threadIdx.x & 31;
    const int wave = threadIdx.x >> 5;
    const int m16  = lane & 15;          // A: M row / B: column
    const int gb   = (lane >> 4) * 8;    // K-group offset (and C/D row +8 group)

    const _Float16* vtb = Vt + (size_t)b * NPOS * CCH;

    Frag16 a[4];
    {
        const _Float16* ap = vtb + (size_t)(n0 + m16) * CCH + gb;
        #pragma unroll
        for (int kc = 0; kc < 4; ++kc) {
            const uint4* p = (const uint4*)(ap + kc * 32);
            a[kc].q[0] = p[0];           // K = kc*32 + gb .. +7
            a[kc].q[1] = p[2];           // K = kc*32 + 16 + gb .. +7
        }
    }

    float* eb = energy + (size_t)b * NPOS * NPOS;

    for (int t = 0; t < 32; ++t) {
        const int m0 = (wave * 32 + t) * 16;
        const _Float16* bp = vtb + (size_t)(m0 + m16) * CCH + gb;
        v8f acc = {};
        #pragma unroll
        for (int kc = 0; kc < 4; ++kc) {
            Frag16 bf;
            const uint4* p = (const uint4*)(bp + kc * 32);
            bf.q[0] = p[0];
            bf.q[1] = p[2];
            acc = __builtin_amdgcn_wmma_f32_16x16x32_f16(
                false, a[kc].h, false, bf.h, (short)0, acc, false, false);
        }
        #pragma unroll
        for (int r = 0; r < 8; ++r) {    // C/D layout: VGPR r -> M = r (+8 hi half)
            eb[(size_t)(n0 + r + gb) * NPOS + m0 + m16] = acc[r];
        }
    }
}

// ---------------------------------------------------------------------------
// K3: in-place row softmax over the 537 MB attention slab.
// One workgroup per (b, row); LDS tree reduction (wave32-safe, no lane-mask
// assumptions). Streaming read+write, single pass of exp applied twice
// (recompute is cheaper than staging 16 KB rows).
// ---------------------------------------------------------------------------
__global__ void pam_row_softmax(float* __restrict__ att) {
    const int n = blockIdx.x;
    const int b = blockIdx.y;
    float* row = att + ((size_t)b * NPOS + n) * NPOS;
    __shared__ float sm[256];
    const int t = threadIdx.x;

    float m = -3.4e38f;
    for (int i = t; i < NPOS; i += 256) m = fmaxf(m, row[i]);
    sm[t] = m; __syncthreads();
    for (int off = 128; off > 0; off >>= 1) {
        if (t < off) sm[t] = fmaxf(sm[t], sm[t + off]);
        __syncthreads();
    }
    m = sm[0]; __syncthreads();

    float s = 0.f;
    for (int i = t; i < NPOS; i += 256) s += __expf(row[i] - m);
    sm[t] = s; __syncthreads();
    for (int off = 128; off > 0; off >>= 1) {
        if (t < off) sm[t] += sm[t + off];
        __syncthreads();
    }
    const float inv = 1.f / sm[0];
    for (int i = t; i < NPOS; i += 256) row[i] = __expf(row[i] - m) * inv;
}

// ---------------------------------------------------------------------------
// K4: out = gamma * (V . A^T) + xs.
// Workgroup = (b, 16-col stripe n0) covering ALL 128 c-rows: 8 waves, wave w
// owns c-rows w*16..+15. Each 16(col) x 32(K=m) attention chunk is converted
// f32->f16 into LDS ONCE and consumed by all 8 waves -> attention is read from
// HBM exactly once (537 MB). A fragments stream from L2-resident V.
// ---------------------------------------------------------------------------
__global__ void pam_out(const _Float16* __restrict__ V,
                        const float* __restrict__ att,
                        const float* __restrict__ gamma,
                        float* __restrict__ out) {
    const int b    = blockIdx.y;
    const int n0   = blockIdx.x * 16;
    const int lane = threadIdx.x & 31;
    const int wave = threadIdx.x >> 5;
    const int m16  = lane & 15;
    const int gb   = (lane >> 4) * 8;
    const int c0   = wave * 16;

    __shared__ _Float16 lb[16 * 32];     // [col][k] f16 staging, 1 KB

    const float*    attb = att + (size_t)b * NPOS * NPOS + (size_t)n0 * NPOS;
    const _Float16* arow = V + ((size_t)b * CCH + c0 + m16) * NPOS + gb;

    v8f acc = {};
    for (int k0 = 0; k0 < NPOS; k0 += 32) {
        __syncthreads();
        for (int i = threadIdx.x; i < 512; i += 256) {
            const int cc = i >> 5, kk = i & 31;
            lb[i] = (_Float16)attb[(size_t)cc * NPOS + k0 + kk];
        }
        __syncthreads();

        Frag16 af, bf;
        const uint4* ap = (const uint4*)(arow + k0);
        af.q[0] = ap[0];
        af.q[1] = ap[2];
        if (k0 + 32 < NPOS)              // pull next A line (global_prefetch_b8)
            __builtin_prefetch((const void*)(arow + k0 + 32), 0, 3);
        const uint4* bp = (const uint4*)(lb + m16 * 32 + gb);
        bf.q[0] = bp[0];
        bf.q[1] = bp[2];

        acc = __builtin_amdgcn_wmma_f32_16x16x32_f16(
            false, af.h, false, bf.h, (short)0, acc, false, false);
    }

    const float g = gamma[0];
    #pragma unroll
    for (int r = 0; r < 8; ++r) {
        const int c = c0 + r + gb;
        const size_t idx = ((size_t)b * CCH + c) * NPOS + n0 + m16;
        out[idx] = g * acc[r] + out[idx];   // out currently holds xs from K1
    }
}

// ---------------------------------------------------------------------------
extern "C" void kernel_launch(void* const* d_in, const int* in_sizes, int n_in,
                              void* d_out, int out_size, void* d_ws, size_t ws_size,
                              hipStream_t stream) {
    (void)in_sizes; (void)n_in; (void)out_size; (void)ws_size;
    const float* x     = (const float*)d_in[0];
    const float* gamma = (const float*)d_in[1];

    float* out = (float*)d_out;                               // B*C*N f32 (xs then final)
    float* att = out + (size_t)BATCH * CCH * NPOS;            // B*N*N f32 (energy->attention)

    _Float16* V  = (_Float16*)d_ws;                           // B*C*N f16 (8 MB)
    _Float16* Vt = V + (size_t)BATCH * CCH * NPOS;            // B*N*C f16 (8 MB)

    pam_channel_softmax<<<dim3(NPOS / 256, BATCH), 256, 0, stream>>>(x, out, V, Vt);
    pam_energy         <<<dim3(NPOS / 16,  BATCH), 256, 0, stream>>>(Vt, att);
    pam_row_softmax    <<<dim3(NPOS,       BATCH), 256, 0, stream>>>(att);
    pam_out            <<<dim3(NPOS / 16,  BATCH), 256, 0, stream>>>(V, att, gamma, out);
}
